// ResidualVQ_45148696216678
// MI455X (gfx1250) — matrix operand order repaired and checked by reading, not compiled, not run on hardware
//
#include <hip/hip_runtime.h>
#include <hip/hip_bf16.h>

// ---------------------------------------------------------------------------
// Residual VQ for MI455X (gfx1250, wave32, WMMA).
// Strategy: distance argmin needs only  cnorm[k] - 2 * <r, c_k>  per row.
// Big GEMM done with v_wmma_f32_16x16x32_bf16 (bf16 inputs, f32 accum).
// Residual update of stage q-1 is fused into the A-tile staging of stage q.
// ---------------------------------------------------------------------------

typedef __attribute__((ext_vector_type(16))) __bf16 v16bf;
typedef __attribute__((ext_vector_type(8)))  float  v8f;
typedef __attribute__((ext_vector_type(4)))  float  f4;     // native vector (NT builtins)

#define RVQ_N 65536
#define RVQ_D 256
#define RVQ_K 1024
#define MT    64            // rows per workgroup in the GEMM
#define LDSR  264           // bf16 elements per LDS row (256 + 8 pad, bank-spread)

__device__ __forceinline__ unsigned short f2bf(float f) {
    unsigned int u = __float_as_uint(f);
    unsigned int r = (u + 0x7FFFu + ((u >> 16) & 1u)) >> 16;   // round-nearest-even
    return (unsigned short)r;
}

// monotonic float -> uint key (ascending)
__device__ __forceinline__ unsigned int fkey(float f) {
    unsigned int u = __float_as_uint(f);
    return (u & 0x80000000u) ? ~u : (u | 0x80000000u);
}

union Frag16 { uint4 q[2]; v16bf v; };

// ---------------------------------------------------------------------------
// prep: optionally run codebook through MLP (cb @ W^T + b), emit bf16 copy
// and per-row squared norms. grid = K blocks, 256 threads (one per out col).
// ---------------------------------------------------------------------------
__global__ __launch_bounds__(256) void rvq_prep(
    const float* __restrict__ cbIn, const float* __restrict__ W,
    const float* __restrict__ bias, float* __restrict__ cbOut,
    unsigned short* __restrict__ cb16, float* __restrict__ cnorm, int doMlp)
{
    __shared__ float rowIn[RVQ_D];
    __shared__ float red[256];
    const int k = blockIdx.x;
    const int i = threadIdx.x;
    float s;
    if (doMlp) {
        rowIn[i] = cbIn[(size_t)k * RVQ_D + i];
        __syncthreads();
        const float* w = W + (size_t)i * RVQ_D;   // out[k][i] = sum_j cb[k][j]*W[i][j]
        s = bias[i];
        #pragma unroll 4
        for (int j = 0; j < RVQ_D; j += 4) {
            float4 wv = *(const float4*)(w + j);
            s += rowIn[j] * wv.x + rowIn[j + 1] * wv.y
               + rowIn[j + 2] * wv.z + rowIn[j + 3] * wv.w;
        }
        cbOut[(size_t)k * RVQ_D + i] = s;
    } else {
        s = cbIn[(size_t)k * RVQ_D + i];
    }
    cb16[(size_t)k * RVQ_D + i] = f2bf(s);
    red[i] = s * s;
    __syncthreads();
    for (int off = 128; off > 0; off >>= 1) {
        if (i < off) red[i] += red[i + off];
        __syncthreads();
    }
    if (i == 0) cnorm[k] = red[0];
}

// ---------------------------------------------------------------------------
// Fused (residual update from previous stage) + bf16 GEMM + argmin.
//   dist(n,k) = cnorm[k] - 2 * <r_n, c_k>   (|r_n|^2 dropped: row-constant)
// grid = N/MT blocks, 256 threads (8 waves). Each wave owns 16-column tiles,
// 4 row-tiles of 16 -> v_wmma_f32_16x16x32_bf16 accumulation over D=256.
// ---------------------------------------------------------------------------
__global__ __launch_bounds__(256) void rvq_gemm_argmin(
    const float* __restrict__ src,      // residual source ([N,D]; x for stage 0)
    const float* __restrict__ subCb,    // f32 codebook of stage q-1 (gather-subtract)
    float* __restrict__ resOut,         // residual out [N,D] (may alias src)
    const unsigned short* __restrict__ cb16,  // bf16 codebook of stage q [K,D]
    const float* __restrict__ cnorm,    // |c_k|^2 of stage q [K]
    unsigned int* __restrict__ idxBuf,  // in: idx of stage q-1, out: idx of stage q
    int useSub)
{
    __shared__ __align__(16) unsigned short Atile[MT * LDSR];   // 33792 B
    __shared__ unsigned long long minTab[MT];

    const int tid = threadIdx.x;

    // ---- stage A tile: apply previous-stage residual update, store bf16 ----
    {
        const int row  = tid >> 2;            // 0..63
        const int dseg = (tid & 3) << 6;      // 0,64,128,192
        const int grow = blockIdx.x * MT + row;
        const size_t gOff = (size_t)grow * RVQ_D + dseg;
        const float* sp = src + gOff;
        float*       rp = resOut + gOff;
        const float* cp = nullptr;
        if (useSub) {
            unsigned int pi = idxBuf[grow];
            cp = subCb + (size_t)pi * RVQ_D + dseg;
        }
        unsigned short* lp = &Atile[row * LDSR + dseg];
        #pragma unroll 4
        for (int j = 0; j < 64; j += 4) {
            float4 v = *(const float4*)(sp + j);
            if (useSub) {
                float4 c = *(const float4*)(cp + j);
                v.x -= c.x; v.y -= c.y; v.z -= c.z; v.w -= c.w;
            }
            *(float4*)(rp + j) = v;
            uint2 h;
            h.x = (unsigned int)f2bf(v.x) | ((unsigned int)f2bf(v.y) << 16);
            h.y = (unsigned int)f2bf(v.z) | ((unsigned int)f2bf(v.w) << 16);
            *(uint2*)(lp + j) = h;
        }
        if (tid < MT) minTab[tid] = ~0ull;
    }
    __syncthreads();

    // ---- WMMA main loop ----
    const int lane = tid & 31;
    const int wave = tid >> 5;      // 0..7
    const int g    = lane >> 4;     // half-wave
    const int m    = lane & 15;     // row-in-tile (A) / column (B,C)

    float        minv[4][8];
    unsigned int mini[4][8];
    #pragma unroll
    for (int t = 0; t < 4; ++t)
        #pragma unroll
        for (int r = 0; r < 8; ++r) { minv[t][r] = __builtin_inff(); mini[t][r] = 0u; }

    for (int it = 0; it < 8; ++it) {
        const int col0 = (it * 8 + wave) * 16;   // 16-column tile base
        v8f acc[4];
        #pragma unroll
        for (int t = 0; t < 4; ++t) acc[t] = (v8f){0.f,0.f,0.f,0.f,0.f,0.f,0.f,0.f};

        // B fragment source: row (col0+m) of bf16 codebook, K runs 8g..8g+7 and 16+8g..16+8g+7
        const unsigned short* bRow = cb16 + (size_t)(col0 + m) * RVQ_D + g * 8;

        #pragma unroll 2
        for (int kk = 0; kk < RVQ_D; kk += 32) {
            Frag16 B;
            B.q[0] = *(const uint4*)(bRow + kk);
            B.q[1] = *(const uint4*)(bRow + kk + 16);
            // 4 consecutive WMMAs share SRC1 (B): set the RB reuse hint on all
            // but the last (ISA §7.12 VOP3P OPSEL_HI[0]). Reuse flags must be
            // literal constants -> manual unroll of the row-tile group.
            #define RVQ_DO_T(t, RB)                                               \
            {                                                                     \
                const unsigned short* aRow =                                      \
                    &Atile[((t) * 16 + m) * LDSR + kk + g * 8];                   \
                Frag16 A;                                                         \
                A.q[0] = *(const uint4*)(aRow);                                   \
                A.q[1] = *(const uint4*)(aRow + 16);                              \
                acc[t] = __builtin_amdgcn_wmma_f32_16x16x32_bf16(                 \
                    false, A.v, false, B.v, (short)0, acc[t], false, RB);         \
            }
            RVQ_DO_T(0, true)
            RVQ_DO_T(1, true)
            RVQ_DO_T(2, true)
            RVQ_DO_T(3, false)
            #undef RVQ_DO_T
        }

        // fold this column tile into the running per-lane minima
        const float cn        = cnorm[col0 + m];
        const unsigned int ci = (unsigned int)(col0 + m);
        #pragma unroll
        for (int t = 0; t < 4; ++t)
            #pragma unroll
            for (int r = 0; r < 8; ++r) {
                float dv = __builtin_fmaf(-2.0f, acc[t][r], cn);
                if (dv < minv[t][r]) { minv[t][r] = dv; mini[t][r] = ci; }
            }
    }

    // ---- reduce: 16 lanes of each half hold the same row, different cols ----
    #pragma unroll
    for (int t = 0; t < 4; ++t) {
        #pragma unroll
        for (int r = 0; r < 8; ++r) {
            float        v = minv[t][r];
            unsigned int i = mini[t][r];
            #pragma unroll
            for (int s = 8; s >= 1; s >>= 1) {
                float        ov = __shfl_xor(v, s, 32);
                unsigned int oi = __shfl_xor(i, s, 32);
                if (ov < v || (ov == v && oi < i)) { v = ov; i = oi; }
            }
            if (m == 0) {
                const int rowLocal = t * 16 + r + 8 * g;
                unsigned long long key =
                    ((unsigned long long)fkey(v) << 32) | (unsigned long long)i;
                atomicMin(&minTab[rowLocal], key);   // ds_min_u64 across 8 waves
            }
        }
    }
    __syncthreads();

    if (tid < MT) {
        unsigned long long e = minTab[tid];
        idxBuf[blockIdx.x * MT + tid] = (unsigned int)(e & 0xFFFFFFFFull);
    }
}

// ---------------------------------------------------------------------------
// finalize: quantized = x - r3 + cb3[idx3]   (one float4 per thread)
// x / r are dead afterwards and out is write-once: non-temporal hints keep
// this 192MB streaming pass from evicting the L2-resident working set.
// ---------------------------------------------------------------------------
__global__ __launch_bounds__(256) void rvq_finalize(
    const float* __restrict__ x, const float* __restrict__ r,
    const float* __restrict__ cb, const unsigned int* __restrict__ idx,
    float* __restrict__ out)
{
    const size_t t   = (size_t)blockIdx.x * 256 + threadIdx.x;
    const size_t row = t >> 6;                 // 64 float4 per 256-wide row
    const int    c   = (int)(t & 63) << 2;
    const unsigned int id = idx[row];
    const size_t o = row * RVQ_D + c;
    f4 xv = __builtin_nontemporal_load((const f4*)(x + o));
    f4 rv = __builtin_nontemporal_load((const f4*)(r + o));
    f4 cv = *(const f4*)(cb + (size_t)id * RVQ_D + c);
    f4 ov = xv - rv + cv;
    __builtin_nontemporal_store(ov, (f4*)(out + o));
}

// ---------------------------------------------------------------------------
extern "C" void kernel_launch(void* const* d_in, const int* in_sizes, int n_in,
                              void* d_out, int out_size, void* d_ws, size_t ws_size,
                              hipStream_t stream) {
    (void)in_sizes; (void)n_in; (void)out_size; (void)ws_size;
    const float* x        = (const float*)d_in[0];
    const float* codebook = (const float*)d_in[1];
    const float* mlp_w    = (const float*)d_in[2];
    const float* mlp_b    = (const float*)d_in[3];
    // d_in[4] = num_quantizers (device scalar); reference fixes it at 4.

    char* ws = (char*)d_ws;
    float*          r     = (float*)ws;                                   // N*D f32
    float*          cbA   = (float*)(ws + (size_t)RVQ_N * RVQ_D * 4);     // K*D f32
    float*          cbB   = cbA + (size_t)RVQ_K * RVQ_D;                  // K*D f32
    unsigned short* cb16  = (unsigned short*)(cbB + (size_t)RVQ_K * RVQ_D);
    float*          cnorm = (float*)(cb16 + (size_t)RVQ_K * RVQ_D);
    unsigned int*   idx   = (unsigned int*)(cnorm + RVQ_K);

    const int gemmBlocks = RVQ_N / MT;   // 1024

    // stage 0: codebook as-is
    rvq_prep<<<RVQ_K, 256, 0, stream>>>(codebook, nullptr, nullptr, nullptr,
                                        cb16, cnorm, 0);
    rvq_gemm_argmin<<<gemmBlocks, 256, 0, stream>>>(x, nullptr, r,
                                                    cb16, cnorm, idx, 0);

    // stages 1..3: MLP-refined codebook; residual update fused into staging
    const float* cbPrev = codebook;
    float* pingpong[3] = { cbA, cbB, cbA };
    for (int q = 1; q < 4; ++q) {
        float* cbOut = pingpong[q - 1];
        rvq_prep<<<RVQ_K, 256, 0, stream>>>(cbPrev, mlp_w, mlp_b, cbOut,
                                            cb16, cnorm, 1);
        rvq_gemm_argmin<<<gemmBlocks, 256, 0, stream>>>(r, cbPrev, r,
                                                        cb16, cnorm, idx, 1);
        cbPrev = cbOut;
    }

    // out = x - r3 + cb3[idx3]
    const int finBlocks = (int)(((size_t)RVQ_N * RVQ_D / 4) / 256);  // 16384
    rvq_finalize<<<finBlocks, 256, 0, stream>>>(x, r, cbPrev, idx, (float*)d_out);
}
